// VectorQuantizer_25924422599192
// MI455X (gfx1250) — compile-verified, matrix operand order checked by reading, not run on hardware
//
#include <hip/hip_runtime.h>
#include <hip/hip_bf16.h>

typedef __bf16 bf16_t;
typedef __attribute__((ext_vector_type(16))) __bf16 v16bf;
typedef __attribute__((ext_vector_type(8)))  __bf16 v8bf;
typedef __attribute__((ext_vector_type(8)))  float  v8f;
typedef __attribute__((ext_vector_type(4)))  unsigned int v4u;
typedef __attribute__((ext_vector_type(8)))  int v8i;
typedef __attribute__((ext_vector_type(4)))  int v4i;

#define N_ROWS 32768
#define E_DIM  512
#define N_E    8192
#define KP     520            // padded LDS row stride (bf16): 1024B data + 16B pad per row

// LDS byte layout for the GEMM kernel (dynamic LDS only => base offset 0):
#define LDS_ZHI   0u
#define LDS_ZLO   (LDS_ZHI + 64u * KP * 2u)         //  66,560
#define LDS_WBUF  (LDS_ZLO + 64u * KP * 2u)         // 133,120
#define WBUF_SZ   (2u * 32u * KP * 2u)              //  66,560 per ping-pong buffer (hi+lo)
#define LDS_RED   (LDS_WBUF + 2u * WBUF_SZ)         // 266,240
#define LDS_TOTAL (LDS_RED + 128u * 4u + 128u * 4u) // 267,264

// ---------------------------------------------------------------------------
// Kernel 1: split W into bf16 hi/lo, compute ||w||^2 per code, zero the loss.
// ---------------------------------------------------------------------------
__global__ __launch_bounds__(256)
void vq_prep_kernel(const float* __restrict__ W,
                    float* __restrict__ wnorm,
                    bf16_t* __restrict__ Whi,
                    bf16_t* __restrict__ Wlo,
                    float* __restrict__ loss_out) {
    __shared__ float part[8];
    const int e = blockIdx.x;
    const int t = threadIdx.x;

    const float2 w2 = ((const float2*)(W + (size_t)e * E_DIM))[t];
    float s = w2.x * w2.x + w2.y * w2.y;

    bf16_t hx = (bf16_t)w2.x;
    bf16_t hy = (bf16_t)w2.y;
    Whi[(size_t)e * E_DIM + 2 * t]     = hx;
    Whi[(size_t)e * E_DIM + 2 * t + 1] = hy;
    Wlo[(size_t)e * E_DIM + 2 * t]     = (bf16_t)(w2.x - (float)hx);
    Wlo[(size_t)e * E_DIM + 2 * t + 1] = (bf16_t)(w2.y - (float)hy);

    #pragma unroll
    for (int off = 16; off > 0; off >>= 1) s += __shfl_xor(s, off, 32);
    if ((t & 31) == 0) part[t >> 5] = s;
    __syncthreads();
    if (t == 0) {
        float tot = 0.f;
        #pragma unroll
        for (int i = 0; i < 8; ++i) tot += part[i];
        wnorm[e] = tot;
        if (e == 0) *loss_out = 0.0f;   // stream-ordered before kernel 3's atomics
    }
}

// ---------------------------------------------------------------------------
// TDM: one descriptor loads a 32x512 bf16 tile (rows of 1024B) into LDS with
// hardware padding of 4 DWORDs after every 256 DWORDs  =>  KP=520 row stride.
// ---------------------------------------------------------------------------
__device__ __forceinline__
void tdm_load_tile(const bf16_t* __restrict__ gsrc, int nBase, unsigned ldsByte) {
    unsigned long long ga = (unsigned long long)(uintptr_t)gsrc
                          + (unsigned long long)nBase * (E_DIM * 2);
    v4u g0;
    g0[0] = 1u;                                             // count=1 (valid), user D#
    g0[1] = ldsByte;                                        // lds_addr
    g0[2] = (unsigned)(ga & 0xFFFFFFFFu);                   // global_addr[31:0]
    g0[3] = (unsigned)((ga >> 32) & 0x01FFFFFFu)            // global_addr[56:32]
          | (2u << 30);                                     // type = 2 ("image")
    v8i g1;
    g1[0] = (1 << 16)        // data_size = 1 -> 2 bytes
          | (1 << 20)        // pad_enable
          | (7 << 22)        // pad_interval: 256 DWORDs (one 1024B row)
          | (3 << 25);       // pad_amount:   4 DWORDs (16B)  => KP row stride
    g1[1] = (int)(512u << 16);   // tensor_dim0 = 512 elements (low 16 bits)
    g1[2] = (int)(8192u << 16);  // tensor_dim1 = 8192 rows (low 16 bits)
    g1[3] = (int)(512u << 16);   // tile_dim0 = 512
    g1[4] = 32;                  // tile_dim1 = 32 rows, tile_dim2 = 0
    g1[5] = 512;                 // tensor_dim0_stride = 512 elements
    g1[6] = 0;
    g1[7] = 0;
    v4i zz4 = {0, 0, 0, 0};          // 2-D tensor: groups 2/3 unused
    v8i zz8 = {0, 0, 0, 0, 0, 0, 0, 0};
    // 6-arg form (clang-23 / therock-10.0 headers)
    __builtin_amdgcn_tensor_load_to_lds(g0, g1, zz4, zz4, zz8, 0);
}

// ---------------------------------------------------------------------------
// Kernel 2: fused (z . W^T) GEMM via compensated-bf16 WMMA + running argmin of
//           (||w||^2 - 2 z.w).  64 rows per WG, all 8192 codes, TDM-ping-pong
//           W tiles (wave 0 drives the Tensor Data Mover).
// ---------------------------------------------------------------------------
__global__ __launch_bounds__(256)
void vq_argmin_kernel(const float* __restrict__ z,
                      const bf16_t* __restrict__ Whi,
                      const bf16_t* __restrict__ Wlo,
                      const float* __restrict__ wnorm,
                      float* __restrict__ idx_out_f,
                      int* __restrict__ idx_out_i) {
    extern __shared__ char smem[];
    bf16_t* zhi    = (bf16_t*)(smem + LDS_ZHI);
    bf16_t* zlo    = (bf16_t*)(smem + LDS_ZLO);
    float*  redMin = (float*)(smem + LDS_RED);
    int*    redIdx = (int*)(smem + LDS_RED + 128 * 4);

    const int tid   = threadIdx.x;
    const int lane  = tid & 31;
    const int wave  = tid >> 5;
    const int waveM = wave & 3;     // 4 tiles along M
    const int waveN = wave >> 2;    // 2 tiles along N
    const int rowBase = blockIdx.x * 64;

    const int nLane = lane & 15;
    const int koff  = (lane < 16) ? 0 : 8;      // ISA 16-bit A/B lane K-offset
    const int mRow  = waveM * 16 + nLane;       // A row owned by this lane

    // prefetch tile 0 via TDM while we stage z
    if (wave == 0) {
        tdm_load_tile(Whi, 0, LDS_WBUF);
        tdm_load_tile(Wlo, 0, LDS_WBUF + 32u * KP * 2u);
    }

    // ---- stage the 64x512 z tile, splitting f32 -> bf16 hi + lo ----
    #pragma unroll 4
    for (int t = 0; t < 32; ++t) {
        int chunk = tid + t * 256;              // 8192 float4 chunks
        int r  = chunk >> 7;                    // 128 chunks / row
        int c4 = (chunk & 127) * 4;
        float4 v = ((const float4*)(z + (size_t)(rowBase + r) * E_DIM))[c4 >> 2];
        float xs[4] = {v.x, v.y, v.z, v.w};
        #pragma unroll
        for (int j = 0; j < 4; ++j) {
            bf16_t h = (bf16_t)xs[j];
            zhi[r * KP + c4 + j] = h;
            zlo[r * KP + c4 + j] = (bf16_t)(xs[j] - (float)h);
        }
    }

    float runMin[8];
    int   runIdx[8];
    #pragma unroll
    for (int i = 0; i < 8; ++i) { runMin[i] = 3.4e38f; runIdx[i] = 0; }

    for (int it = 0; it < N_E / 32; ++it) {
        const int nBase = it * 32;
        const unsigned curBuf = LDS_WBUF + (unsigned)(it & 1) * WBUF_SZ;
        const unsigned nxtBuf = LDS_WBUF + (unsigned)((it + 1) & 1) * WBUF_SZ;

        if (wave == 0) {
            if (it + 1 < N_E / 32) {
                tdm_load_tile(Whi, nBase + 32, nxtBuf);
                tdm_load_tile(Wlo, nBase + 32, nxtBuf + 32u * KP * 2u);
                __builtin_amdgcn_s_wait_tensorcnt((short)2);  // current tile landed
            } else {
                __builtin_amdgcn_s_wait_tensorcnt((short)0);
            }
        }
        __syncthreads();   // publish current tile to all waves

        const bf16_t* whiL = (const bf16_t*)(smem + curBuf);
        const bf16_t* wloL = whiL + 32 * KP;

        // three independent accumulator chains (hi*hi, hi*lo, lo*hi)
        v8f chh = {0.f,0.f,0.f,0.f,0.f,0.f,0.f,0.f};
        v8f chl = {0.f,0.f,0.f,0.f,0.f,0.f,0.f,0.f};
        v8f clh = {0.f,0.f,0.f,0.f,0.f,0.f,0.f,0.f};

        const bf16_t* aHiRow = zhi  + mRow * KP;
        const bf16_t* aLoRow = zlo  + mRow * KP;
        const bf16_t* bHiRow = whiL + (waveN * 16 + nLane) * KP;
        const bf16_t* bLoRow = wloL + (waveN * 16 + nLane) * KP;

        #pragma unroll 4
        for (int k0 = 0; k0 < E_DIM; k0 += 32) {
            v8bf ah0 = *(const v8bf*)(aHiRow + k0 + koff);
            v8bf ah1 = *(const v8bf*)(aHiRow + k0 + 16 + koff);
            v8bf al0 = *(const v8bf*)(aLoRow + k0 + koff);
            v8bf al1 = *(const v8bf*)(aLoRow + k0 + 16 + koff);
            v8bf bh0 = *(const v8bf*)(bHiRow + k0 + koff);
            v8bf bh1 = *(const v8bf*)(bHiRow + k0 + 16 + koff);
            v8bf bl0 = *(const v8bf*)(bLoRow + k0 + koff);
            v8bf bl1 = *(const v8bf*)(bLoRow + k0 + 16 + koff);
            v16bf aH = __builtin_shufflevector(ah0, ah1, 0,1,2,3,4,5,6,7,8,9,10,11,12,13,14,15);
            v16bf aL = __builtin_shufflevector(al0, al1, 0,1,2,3,4,5,6,7,8,9,10,11,12,13,14,15);
            v16bf bH = __builtin_shufflevector(bh0, bh1, 0,1,2,3,4,5,6,7,8,9,10,11,12,13,14,15);
            v16bf bL = __builtin_shufflevector(bl0, bl1, 0,1,2,3,4,5,6,7,8,9,10,11,12,13,14,15);
            chh = __builtin_amdgcn_wmma_f32_16x16x32_bf16(false, aH, false, bH, (short)0, chh, false, false);
            chl = __builtin_amdgcn_wmma_f32_16x16x32_bf16(false, aH, false, bL, (short)0, chl, false, false);
            clh = __builtin_amdgcn_wmma_f32_16x16x32_bf16(false, aL, false, bH, (short)0, clh, false, false);
        }

        // ---- epilogue: running per-lane argmin of (||w||^2 - 2 z.w) ----
        const int n = nBase + waveN * 16 + nLane;   // C/D: N = lane&15
        const float wn = wnorm[n];
        #pragma unroll
        for (int i = 0; i < 8; ++i) {
            float dot = chh[i] + (chl[i] + clh[i]);
            float dp  = fmaf(-2.0f, dot, wn);
            if (dp < runMin[i]) { runMin[i] = dp; runIdx[i] = n; }
        }
        __syncthreads();   // done reading curBuf; TDM may overwrite it next iter
    }

    // ---- reduce across the 16 lanes holding each C row, then across waveN ----
    #pragma unroll
    for (int i = 0; i < 8; ++i) {
        float m = runMin[i]; int ix = runIdx[i];
        #pragma unroll
        for (int off = 1; off < 16; off <<= 1) {
            float om = __shfl_xor(m, off, 32);
            int   oi = __shfl_xor(ix, off, 32);
            if (om < m || (om == m && oi < ix)) { m = om; ix = oi; }
        }
        if (nLane == 0) {
            // C/D layout: VGPR i -> row i (lanes 0-15) or row i+8 (lanes 16-31)
            int rloc = waveM * 16 + i + ((lane & 16) ? 8 : 0);
            redMin[waveN * 64 + rloc] = m;
            redIdx[waveN * 64 + rloc] = ix;
        }
    }
    __syncthreads();
    if (tid < 64) {
        float m0 = redMin[tid], m1 = redMin[64 + tid];
        int   i0 = redIdx[tid], i1 = redIdx[64 + tid];
        int best = (m1 < m0 || (m1 == m0 && i1 < i0)) ? i1 : i0;
        idx_out_i[rowBase + tid] = best;
        idx_out_f[rowBase + tid] = (float)best;
    }
}

// ---------------------------------------------------------------------------
// Kernel 3: z_q = W[idx] (exact f32 gather) and loss = 2 * mean((z_q - z)^2).
// ---------------------------------------------------------------------------
__global__ __launch_bounds__(256)
void vq_gather_loss_kernel(const float* __restrict__ z,
                           const float* __restrict__ W,
                           const int* __restrict__ idx,
                           float* __restrict__ zq_out,
                           float* __restrict__ loss_out) {
    __shared__ float part[8];
    const int m = blockIdx.x;
    const int t = threadIdx.x;
    const int e = idx[m];

    const float2 w2 = ((const float2*)(W + (size_t)e * E_DIM))[t];
    const float2 z2 = ((const float2*)(z + (size_t)m * E_DIM))[t];
    ((float2*)(zq_out + (size_t)m * E_DIM))[t] = w2;

    float dx = w2.x - z2.x, dy = w2.y - z2.y;
    float s = dx * dx + dy * dy;
    #pragma unroll
    for (int off = 16; off > 0; off >>= 1) s += __shfl_xor(s, off, 32);
    if ((t & 31) == 0) part[t >> 5] = s;
    __syncthreads();
    if (t == 0) {
        float tot = 0.f;
        #pragma unroll
        for (int i = 0; i < 8; ++i) tot += part[i];
        // loss = mean(diff^2) + BETA * mean(diff^2), BETA = 1
        atomicAdd(loss_out, 2.0f * tot / ((float)N_ROWS * (float)E_DIM));
    }
}

// ---------------------------------------------------------------------------
extern "C" void kernel_launch(void* const* d_in, const int* in_sizes, int n_in,
                              void* d_out, int out_size, void* d_ws, size_t ws_size,
                              hipStream_t stream) {
    (void)in_sizes; (void)n_in; (void)out_size; (void)ws_size;

    const float* z = (const float*)d_in[0];           // 32768 x 512 (x1x1) f32
    const float* W = (const float*)d_in[1];           // 8192 x 512 f32

    // d_out (f32): [loss(1)] [z_q (32768*512)] [indices-as-float (32768)]
    float* out  = (float*)d_out;
    float* loss = out;
    float* zq   = out + 1;
    float* idxf = out + 1 + (size_t)N_ROWS * E_DIM;

    // workspace: wnorm | Whi | Wlo | idx(int)   (~16.2 MB)
    char* ws = (char*)d_ws;
    float*  wnorm = (float*)ws;
    bf16_t* Whi   = (bf16_t*)(ws + 32768);
    bf16_t* Wlo   = (bf16_t*)(ws + 32768 + (size_t)N_E * E_DIM * 2);
    int*    idxi  = (int*)  (ws + 32768 + 2 * (size_t)N_E * E_DIM * 2);

    const size_t smem = LDS_TOTAL;   // ~261 KB < 320 KB/WGP
    (void)hipFuncSetAttribute((const void*)vq_argmin_kernel,
                              hipFuncAttributeMaxDynamicSharedMemorySize, (int)smem);

    vq_prep_kernel<<<N_E, 256, 0, stream>>>(W, wnorm, Whi, Wlo, loss);
    vq_argmin_kernel<<<N_ROWS / 64, 256, smem, stream>>>(z, Whi, Wlo, wnorm, idxf, idxi);
    vq_gather_loss_kernel<<<N_ROWS, 256, 0, stream>>>(z, W, idxi, zq, loss);
}
